// BidirectionalInfoNCELoss_35923106463793
// MI455X (gfx1250) — matrix operand order repaired and checked by reading, not compiled, not run on hardware
//
#include <hip/hip_runtime.h>
#include <hip/hip_bf16.h>
#include <math.h>

// ---------------------------------------------------------------------------
// Bidirectional InfoNCE (CLAP-style) on MI455X / gfx1250.
//   sim = normalize(A) . normalize(T)^T / temp        (8192 x 8192)
//   loss = 0.5*(mean_i LSE_row_i + mean_j LSE_col_j) - mean_i sim[i,i]
//
// bf16 WMMA GEMM (v_wmma_f32_16x16x32_bf16) with fp32 accumulate.
// |sim| <= 1/0.07 ~ 14.3 => exp() never overflows fp32 when summed over 8192
// columns (<= 1.4e10), so NO online max tracking is required: each wave just
// accumulates sum(exp(sim)) per row and takes one log at the end.
// 1/temp is folded into the audio side at normalization time, so it applies
// to the whole product in BOTH gemm passes and to the diagonal.
//
// Tiling: block = 256 threads (8 wave32) owns 32 query rows. Each wave
// computes a 32x64 strip per column iteration: 2 A fragments x 4 B fragments
// -> 8 v_wmma per K-step with only 12 b128 loads (B reused x2, A x4).
// No LDS staging: the 32 shared A rows (64KB) sit in WGP$/L0; the 16MB bf16
// key matrix sits in the 192MB L2, so HBM traffic is negligible and the
// kernel is pure WMMA-throughput bound.
// ---------------------------------------------------------------------------

typedef __attribute__((ext_vector_type(16))) __bf16 v16bf;
typedef __attribute__((ext_vector_type(8)))  __bf16 v8bf;
typedef __attribute__((ext_vector_type(8)))  float  v8f;

#define NB      8192          // batch
#define ND      1024          // dim
#define INV_T   (1.0f / 0.07f)

// -------------------------------------------------------------------- utils
__device__ __forceinline__ v16bf load_frag(const __bf16* p0, const __bf16* p1) {
    v8bf lo = *(const v8bf*)p0;
    v8bf hi = *(const v8bf*)p1;
    v16bf r;
#pragma unroll
    for (int i = 0; i < 8; ++i) { r[i] = lo[i]; r[i + 8] = hi[i]; }
    return r;
}

// --------------------------------------------------- 1) normalize -> bf16
// One block per row (2*NB blocks). Audio rows additionally scaled by 1/temp.
__global__ __launch_bounds__(256)
void normalize_bf16_kernel(const float* __restrict__ audio,
                           const float* __restrict__ text,
                           unsigned short* __restrict__ outA,
                           unsigned short* __restrict__ outT) {
    __shared__ float red[256];
    const int row     = blockIdx.x;
    const bool isA    = row < NB;
    const int r       = isA ? row : row - NB;
    const float* src  = (isA ? audio : text) + (size_t)r * ND;
    unsigned short* dst = (isA ? outA : outT) + (size_t)r * ND;

    const int t = threadIdx.x;
    float4 v = ((const float4*)src)[t];
    red[t] = v.x * v.x + v.y * v.y + v.z * v.z + v.w * v.w;
    __syncthreads();
    for (int off = 128; off > 0; off >>= 1) {
        if (t < off) red[t] += red[t + off];
        __syncthreads();
    }
    const float scale = rsqrtf(red[0]) * (isA ? INV_T : 1.0f);

    __bf16 o0 = (__bf16)(v.x * scale);
    __bf16 o1 = (__bf16)(v.y * scale);
    __bf16 o2 = (__bf16)(v.z * scale);
    __bf16 o3 = (__bf16)(v.w * scale);
    dst[4 * t + 0] = __builtin_bit_cast(unsigned short, o0);
    dst[4 * t + 1] = __builtin_bit_cast(unsigned short, o1);
    dst[4 * t + 2] = __builtin_bit_cast(unsigned short, o2);
    dst[4 * t + 3] = __builtin_bit_cast(unsigned short, o3);
}

// ----------------------------------------- 2) fused GEMM + row logsumexp
// Grid = NB/32 blocks. Wave w covers columns [w*64 + n*512, +64).
__global__ __launch_bounds__(256)
void lse_rows_kernel(const unsigned short* __restrict__ Qus,
                     const unsigned short* __restrict__ Kus,
                     float* __restrict__ outLse) {
    const __bf16* Q  = (const __bf16*)Qus;
    const __bf16* Kd = (const __bf16*)Kus;

    __shared__ float lsePart[32];

    const int tid     = threadIdx.x;
    const int rowbase = blockIdx.x * 32;
    if (tid < 32) lsePart[tid] = 0.0f;
    __syncthreads();

    const int lane = tid & 31;
    const int wave = tid >> 5;
    const int half = lane >> 4;   // which 16-lane half
    const int lm   = lane & 15;

    // A fragment (16x32 bf16): lane m -> K[k..k+15] ; lane m+16 -> K[k+16..k+31]
    const int aOff = half * 16;         // + ks ; second chunk +8
    // B fragment (32x16 bf16): lane n -> K[k..k+15] ; lane n+16 -> K[k+16..k+31]
    const int bOff = half * 16;         // + ks ; second chunk +8

    const __bf16* a0 = Q + (size_t)(rowbase +  0 + lm) * ND;  // rows 0..15
    const __bf16* a1 = Q + (size_t)(rowbase + 16 + lm) * ND;  // rows 16..31

    v8f se0 = {};   // running sum(exp) for rows 0..15 of the block
    v8f se1 = {};   // running sum(exp) for rows 16..31

    for (int cb = wave * 64; cb < NB; cb += 8 * 64) {
        v8f acc00 = {}, acc01 = {}, acc02 = {}, acc03 = {};
        v8f acc10 = {}, acc11 = {}, acc12 = {}, acc13 = {};
        const __bf16* b0 = Kd + (size_t)(cb +  0 + lm) * ND;
        const __bf16* b1 = Kd + (size_t)(cb + 16 + lm) * ND;
        const __bf16* b2 = Kd + (size_t)(cb + 32 + lm) * ND;
        const __bf16* b3 = Kd + (size_t)(cb + 48 + lm) * ND;

#pragma unroll 2
        for (int ks = 0; ks < ND; ks += 32) {
            v16bf af0 = load_frag(a0 + ks + aOff, a0 + ks + aOff + 8);
            v16bf af1 = load_frag(a1 + ks + aOff, a1 + ks + aOff + 8);

            v16bf bf0 = load_frag(b0 + ks + bOff, b0 + ks + bOff + 8);
            acc00 = __builtin_amdgcn_wmma_f32_16x16x32_bf16(false, af0, false, bf0,
                                                            (short)0, acc00, false, false);
            acc10 = __builtin_amdgcn_wmma_f32_16x16x32_bf16(false, af1, false, bf0,
                                                            (short)0, acc10, false, false);
            v16bf bf1 = load_frag(b1 + ks + bOff, b1 + ks + bOff + 8);
            acc01 = __builtin_amdgcn_wmma_f32_16x16x32_bf16(false, af0, false, bf1,
                                                            (short)0, acc01, false, false);
            acc11 = __builtin_amdgcn_wmma_f32_16x16x32_bf16(false, af1, false, bf1,
                                                            (short)0, acc11, false, false);
            v16bf bf2 = load_frag(b2 + ks + bOff, b2 + ks + bOff + 8);
            acc02 = __builtin_amdgcn_wmma_f32_16x16x32_bf16(false, af0, false, bf2,
                                                            (short)0, acc02, false, false);
            acc12 = __builtin_amdgcn_wmma_f32_16x16x32_bf16(false, af1, false, bf2,
                                                            (short)0, acc12, false, false);
            v16bf bf3 = load_frag(b3 + ks + bOff, b3 + ks + bOff + 8);
            acc03 = __builtin_amdgcn_wmma_f32_16x16x32_bf16(false, af0, false, bf3,
                                                            (short)0, acc03, false, false);
            acc13 = __builtin_amdgcn_wmma_f32_16x16x32_bf16(false, af1, false, bf3,
                                                            (short)0, acc13, false, false);
        }
        // Temperature already folded in; just exp-accumulate.
#pragma unroll
        for (int i = 0; i < 8; ++i) {
            se0[i] += __expf(acc00[i]) + __expf(acc01[i]) +
                      __expf(acc02[i]) + __expf(acc03[i]);
            se1[i] += __expf(acc10[i]) + __expf(acc11[i]) +
                      __expf(acc12[i]) + __expf(acc13[i]);
        }
    }

    // Row sums: N lives across the 16 lanes of each half.
    // C layout: (lanes 0-15, vgpr r) -> row r ; (lanes 16-31, vgpr r) -> row r+8.
#pragma unroll
    for (int m = 1; m < 16; m <<= 1)
#pragma unroll
        for (int i = 0; i < 8; ++i) {
            se0[i] += __shfl_xor(se0[i], m, 32);
            se1[i] += __shfl_xor(se1[i], m, 32);
        }

    if (lm == 0) {
        const int ro = half * 8;
#pragma unroll
        for (int r = 0; r < 8; ++r) {
            atomicAdd(&lsePart[ro + r],      se0[r]);
            atomicAdd(&lsePart[16 + ro + r], se1[r]);
        }
    }
    __syncthreads();
    if (tid < 32) outLse[rowbase + tid] = __logf(lsePart[tid]);
}

// ------------------------------------------------ 3) diagonal sim[i,i]
// One wave per row: dot(A'[i], T'[i]) (1/temp already folded into A').
__global__ __launch_bounds__(256)
void diag_kernel(const unsigned short* __restrict__ Aus,
                 const unsigned short* __restrict__ Tus,
                 float* __restrict__ diag) {
    const int gw   = (blockIdx.x * blockDim.x + threadIdx.x) >> 5;
    const int lane = threadIdx.x & 31;
    if (gw >= NB) return;
    const __bf16* a = (const __bf16*)Aus + (size_t)gw * ND;
    const __bf16* t = (const __bf16*)Tus + (size_t)gw * ND;
    float s = 0.0f;
    for (int c = lane; c < ND / 8; c += 32) {
        v8bf va = ((const v8bf*)a)[c];
        v8bf vt = ((const v8bf*)t)[c];
#pragma unroll
        for (int i = 0; i < 8; ++i) s += (float)va[i] * (float)vt[i];
    }
#pragma unroll
    for (int m = 16; m > 0; m >>= 1) s += __shfl_xor(s, m, 32);
    if (lane == 0) diag[gw] = s;
}

// ------------------------------------------------ 4) final scalar loss
__global__ __launch_bounds__(256)
void final_loss_kernel(const float* __restrict__ lseA,
                       const float* __restrict__ lseT,
                       const float* __restrict__ diag,
                       float* __restrict__ out) {
    __shared__ float red[256];
    float s = 0.0f;
    for (int i = threadIdx.x; i < NB; i += 256)
        s += 0.5f * (lseA[i] + lseT[i]) - diag[i];
    red[threadIdx.x] = s;
    __syncthreads();
    for (int off = 128; off > 0; off >>= 1) {
        if (threadIdx.x < off) red[threadIdx.x] += red[threadIdx.x + off];
        __syncthreads();
    }
    if (threadIdx.x == 0) out[0] = red[0] / (float)NB;
}

// ---------------------------------------------------------------- launch
extern "C" void kernel_launch(void* const* d_in, const int* in_sizes, int n_in,
                              void* d_out, int out_size, void* d_ws, size_t ws_size,
                              hipStream_t stream) {
    (void)in_sizes; (void)n_in; (void)out_size; (void)ws_size;
    const float* audio = (const float*)d_in[0];
    const float* text  = (const float*)d_in[1];

    char* ws = (char*)d_ws;
    unsigned short* Abf = (unsigned short*)ws;                               // 16 MB
    unsigned short* Tbf = (unsigned short*)(ws + (size_t)NB * ND * 2);       // 16 MB
    float* lseA = (float*)(ws + (size_t)NB * ND * 4);                        // 32 KB
    float* lseT = lseA + NB;                                                 // 32 KB
    float* diag = lseT + NB;                                                 // 32 KB
    float* out  = (float*)d_out;

    normalize_bf16_kernel<<<2 * NB, 256, 0, stream>>>(audio, text, Abf, Tbf);
    lse_rows_kernel<<<NB / 32, 256, 0, stream>>>(Abf, Tbf, lseA);  // rows: audio->text
    lse_rows_kernel<<<NB / 32, 256, 0, stream>>>(Tbf, Abf, lseT);  // cols: text->audio
    diag_kernel<<<NB / 8, 256, 0, stream>>>(Abf, Tbf, diag);
    final_loss_kernel<<<1, 256, 0, stream>>>(lseA, lseT, diag, out);
}
